// OperatorFilterLayer_35081292873728
// MI455X (gfx1250) — matrix operand order repaired and checked by reading, not compiled, not run on hardware
//
#include <hip/hip_runtime.h>
#include <hip/hip_bf16.h>

// ---------------------------------------------------------------------------
// out[b,o,d] = sum_{i,k,e} coeff[o,i,k] * x[b,i,e] * W[k,e,d]
// B=32, F_IN=F_OUT=64, D=512, K=64.  All fp32 in/out.
//
// Strategy (MI455X / gfx1250):
//  * Compute-bound (~80 GFLOP of GEMM vs ~73MB unique HBM traffic ~3us).
//  * Convert all operands once to bf16 and run everything through
//    v_wmma_f32_16x16x32_bf16 (f32 accumulate).
//  * W in bf16 is 32MB -> resident in the 192MB L2; the 32 per-batch
//    re-reads never touch HBM.
//  * Operands are pre-swizzled in workspace into the exact per-lane WMMA
//    fragment layouts (ISA 7.12.2), so each fragment is one contiguous
//    32-byte per-lane load in the hot loop.
//  * global_prefetch_b8 one k-step ahead on the W fragment stream hides
//    L2 latency behind the current k's WMMAs.
//
// Workspace layout (bf16 elements):
//   Wws : [k][et][dt][lane][h]  64*16*32*32*16 = 16,777,216   (B-frags of W)
//   Cws : [k][ot][it][lane][h]  64* 4* 2*32*16 =    262,144   (A-frags of C_k)
//   Xws : [b][it][e2][lane][h]  32* 2*32*32*16 =  1,048,576   (B-frags of X_b)
//   total = 36,175,872 bytes (~34.5 MB) of d_ws.
// ---------------------------------------------------------------------------

typedef __attribute__((ext_vector_type(16))) __bf16 v16bf;
typedef __attribute__((ext_vector_type(8)))  __bf16 v8bf;
typedef __attribute__((ext_vector_type(8)))  float  v8f;

#define NB   32
#define FIN  64
#define FOUT 64
#define DD   512
#define KK   64

#define SSTR 520  // padded LDS row stride for S (bank spread + 16B alignment)

// A-matrix (16x32, 16-bit) per-lane K index for half h (ISA 7.12.2 table)
__device__ __forceinline__ int akmap(int lane, int h) {
  int base = (lane < 16) ? 0 : 8;
  return (h < 8) ? (h + base) : (h + 8 + base);
}
// B-matrix (32x16, 16-bit) per-lane K index for half h
__device__ __forceinline__ int bkmap(int lane, int h) {
  return akmap(lane, h);
}

// ---- prep: W[k,e,d] fp32 -> bf16 B-fragments -------------------------------
__global__ __launch_bounds__(256) void prep_w(const float* __restrict__ W,
                                              __bf16* __restrict__ Wws) {
  int t = blockIdx.x * blockDim.x + threadIdx.x;   // 1,048,576 threads
  int lane =  t        & 31;
  int dt   = (t >> 5)  & 31;
  int et   = (t >> 10) & 15;
  int k    =  t >> 14;
  int d = dt * 16 + (lane & 15);
  v16bf v;
#pragma unroll
  for (int h = 0; h < 16; ++h) {
    int e = et * 32 + bkmap(lane, h);
    v[h] = (__bf16)W[((size_t)k * DD + e) * DD + d];
  }
  *(v16bf*)(Wws + (size_t)t * 16) = v;
}

// ---- prep: coeff[o,i,k] fp32 -> bf16 A-fragments of C_k --------------------
__global__ __launch_bounds__(256) void prep_c(const float* __restrict__ C,
                                              __bf16* __restrict__ Cws) {
  int t = blockIdx.x * blockDim.x + threadIdx.x;   // 16,384 threads
  int lane =  t       & 31;
  int it   = (t >> 5) & 1;
  int ot   = (t >> 6) & 3;
  int k    =  t >> 8;
  int o = ot * 16 + (lane & 15);
  v16bf v;
#pragma unroll
  for (int h = 0; h < 16; ++h) {
    int i = it * 32 + akmap(lane, h);
    v[h] = (__bf16)C[((size_t)o * FIN + i) * KK + k];
  }
  *(v16bf*)(Cws + (size_t)t * 16) = v;
}

// ---- prep: x[b,i,e] fp32 -> bf16 B-fragments of X_b ------------------------
__global__ __launch_bounds__(256) void prep_x(const float* __restrict__ X,
                                              __bf16* __restrict__ Xws) {
  int t = blockIdx.x * blockDim.x + threadIdx.x;   // 65,536 threads
  int lane =  t        & 31;
  int e2   = (t >> 5)  & 31;
  int it   = (t >> 10) & 1;
  int b    =  t >> 11;
  int e = e2 * 16 + (lane & 15);
  v16bf v;
#pragma unroll
  for (int h = 0; h < 16; ++h) {
    int i = it * 32 + bkmap(lane, h);
    v[h] = (__bf16)X[((size_t)b * FIN + i) * DD + e];
  }
  *(v16bf*)(Xws + (size_t)t * 16) = v;
}

// ---- main fused kernel -----------------------------------------------------
// grid = (D/128, B); block = 256 (8 wave32).  Each WG owns out[b, 0:64, dblk*128 +: 128].
// Per k:  stage1  S = C_k @ X_b          (64x512, K=64)  -> LDS (bf16)
//         stage2  acc += S @ W_k[:,tile] (64x128, K=512)
__global__ __launch_bounds__(256) void opfilter_main(
    const __bf16* __restrict__ Wws, const __bf16* __restrict__ Cws,
    const __bf16* __restrict__ Xws, float* __restrict__ out) {
  __shared__ __align__(16) __bf16 Xlds[2 * 32 * 32 * 16];  // 64 KB, fragment layout
  __shared__ __align__(16) __bf16 Slds[FOUT * SSTR];       // 65 KB, row-major padded

  const int tid   = threadIdx.x;
  const int lane  = tid & 31;
  const int w     = tid >> 5;
  const int b     = blockIdx.y;
  const int dblk  = blockIdx.x;        // 0..3
  const int ot    = w >> 1;            // this wave's o-tile (0..3)
  const int ehalf = w & 1;             // stage1 e-half / stage2 n-half

  // stage 0: copy pre-swizzled X_b into LDS (4096 x 16B, coalesced)
  {
    const float4* src = (const float4*)(Xws + (size_t)b * 32768);
    float4* dst = (float4*)Xlds;
    for (int i = tid; i < 4096; i += 256) dst[i] = src[i];
  }
  __syncthreads();

  v8f acc[4];
#pragma unroll
  for (int j = 0; j < 4; ++j) acc[j] = {};

  const v16bf* XF = (const v16bf*)Xlds;
  const v16bf* CF = (const v16bf*)Cws;
  const v16bf* WF = (const v16bf*)Wws;

  for (int k = 0; k < KK; ++k) {
    // A-fragments of C_k for this wave's o-tile (L2-resident, 2x32B/lane)
    v16bf a0 = CF[(((size_t)k * 4 + ot) * 2 + 0) * 32 + lane];
    v16bf a1 = CF[(((size_t)k * 4 + ot) * 2 + 1) * 32 + lane];

    // Prefetch next k's W fragment slice for this WG into near cache
    // (128KB = 1024 cachelines of 128B; 4 lines per thread; wrap keeps
    //  addresses valid and control flow uniform).
    {
      int kn = (k + 1) & (KK - 1);
      const char* wbase = (const char*)Wws;
#pragma unroll
      for (int q = 0; q < 4; ++q) {
        int line = tid * 4 + q;            // 0..1023
        int et   = line >> 6;              // 64 lines per et-slice
        size_t off =
            ((((size_t)kn * 16 + et) * 32 + (size_t)dblk * 8) * 32 * 16) * 2 +
            (size_t)(line & 63) * 128;
        __builtin_prefetch(wbase + off, 0, 3);
      }
    }

    __syncthreads();  // previous k's stage-2 reads of Slds are done

    // ---- stage 1: S[ot*16 +: 16, ehalf*256 +: 256] ----
#pragma unroll 4
    for (int e2l = 0; e2l < 16; ++e2l) {
      int e2 = ehalf * 16 + e2l;
      v16bf b0 = XF[(0 * 32 + e2) * 32 + lane];
      v16bf b1 = XF[(1 * 32 + e2) * 32 + lane];
      v8f c = {};
      c = __builtin_amdgcn_wmma_f32_16x16x32_bf16(false, a0, false, b0,
                                                  (short)0, c, false, false);
      c = __builtin_amdgcn_wmma_f32_16x16x32_bf16(false, a1, false, b1,
                                                  (short)0, c, false, false);
      // scatter C-fragment (f32) -> row-major bf16 S in LDS
      int n  = e2 * 16 + (lane & 15);
      int m0 = ot * 16 + ((lane < 16) ? 0 : 8);
#pragma unroll
      for (int r = 0; r < 8; ++r)
        Slds[(m0 + r) * SSTR + n] = (__bf16)c[r];
    }
    __syncthreads();  // S complete

    // ---- stage 2: acc += S @ W_k[:, dblk*128 +: 128] ----
#pragma unroll 2
    for (int et = 0; et < 16; ++et) {
      // A-fragment of S from LDS: two aligned 16B runs per lane
      int o  = ot * 16 + (lane & 15);
      int eb = et * 32 + ((lane < 16) ? 0 : 8);
      const __bf16* p = &Slds[o * SSTR + eb];
      v16bf a;
      *(v8bf*)&a         = *(const v8bf*)p;
      *(((v8bf*)&a) + 1) = *(const v8bf*)(p + 16);
#pragma unroll
      for (int j = 0; j < 4; ++j) {
        int dt = dblk * 8 + ehalf * 4 + j;  // global d-tile 0..31
        v16bf bw = WF[(((size_t)k * 16 + et) * 32 + dt) * 32 + lane];
        acc[j] = __builtin_amdgcn_wmma_f32_16x16x32_bf16(
            false, a, false, bw, (short)0, acc[j], false, false);
      }
    }
  }

  // epilogue: C-fragment layout -> out[b,o,d] fp32
  int m0 = ot * 16 + ((lane < 16) ? 0 : 8);
#pragma unroll
  for (int j = 0; j < 4; ++j) {
    int d = dblk * 128 + (ehalf * 4 + j) * 16 + (lane & 15);
#pragma unroll
    for (int r = 0; r < 8; ++r)
      out[((size_t)b * FOUT + (m0 + r)) * DD + d] = acc[j][r];
  }
}

extern "C" void kernel_launch(void* const* d_in, const int* in_sizes, int n_in,
                              void* d_out, int out_size, void* d_ws, size_t ws_size,
                              hipStream_t stream) {
  (void)in_sizes; (void)n_in; (void)out_size; (void)ws_size;
  const float* x     = (const float*)d_in[0];  // [32,64,512]
  const float* wops  = (const float*)d_in[1];  // [64,512,512]
  const float* coeff = (const float*)d_in[2];  // [64,64,64]
  float* out = (float*)d_out;                  // [32,64,512]

  __bf16* Wws = (__bf16*)d_ws;                 // 16,777,216 el
  __bf16* Cws = Wws + (size_t)16777216;        //    262,144 el
  __bf16* Xws = Cws + (size_t)262144;          //  1,048,576 el

  prep_w<<<4096, 256, 0, stream>>>(wops, Wws);
  prep_c<<<64,   256, 0, stream>>>(coeff, Cws);
  prep_x<<<256,  256, 0, stream>>>(x, Xws);

  dim3 grid(DD / 128, NB);
  opfilter_main<<<grid, 256, 0, stream>>>(Wws, Cws, Xws, out);
}
